// SpiralConv_2800318677118
// MI455X (gfx1250) — matrix operand order repaired and checked by reading, not compiled, not run on hardware
//
#include <hip/hip_runtime.h>

typedef _Float16 f16;
typedef __attribute__((ext_vector_type(16))) _Float16 v16h;
typedef __attribute__((ext_vector_type(8)))  float    v8f;

namespace {
constexpr int kNpts   = 12000;   // N points
constexpr int kF      = 128;     // features
constexpr int kS      = 16;      // spiral size
constexpr int kK      = kF * kS; // 2048 (GEMM K)
constexpr int kOut    = 128;     // output channels
constexpr int kKC     = 128;     // K-chunk == one gathered segment
constexpr int kNChunk = kK / kKC;// 16 chunks
constexpr int kRowStr = 136;     // LDS row stride in halves (padded, 16B-aligned)

struct U32x8 { uint4 a, b; };
__device__ __forceinline__ v16h mk16h(uint4 a, uint4 b) {
  U32x8 t{a, b};
  return __builtin_bit_cast(v16h, t);
}
} // namespace

// ---- one-shot prep: split W (fp32) into hi/lo f16 planes in workspace ----
__global__ __launch_bounds__(256) void wsplit_kernel(
    const float* __restrict__ Wm,
    unsigned short* __restrict__ whi, unsigned short* __restrict__ wlo) {
  const int i = (blockIdx.x * 256 + threadIdx.x) * 4;   // 65536 threads * 4
  const float4 v = *(const float4*)(Wm + i);
  const float vv[4] = {v.x, v.y, v.z, v.w};
  unsigned int h01 = 0, h23 = 0, l01 = 0, l23 = 0;
  f16 h[4], l[4];
  #pragma unroll
  for (int j = 0; j < 4; ++j) {
    h[j] = (f16)vv[j];
    l[j] = (f16)(vv[j] - (float)h[j]);
  }
  unsigned short hb[4], lb[4];
  __builtin_memcpy(hb, h, 8);
  __builtin_memcpy(lb, l, 8);
  h01 = (unsigned int)hb[0] | ((unsigned int)hb[1] << 16);
  h23 = (unsigned int)hb[2] | ((unsigned int)hb[3] << 16);
  l01 = (unsigned int)lb[0] | ((unsigned int)lb[1] << 16);
  l23 = (unsigned int)lb[2] | ((unsigned int)lb[3] << 16);
  *(uint2*)(whi + i) = make_uint2(h01, h23);
  *(uint2*)(wlo + i) = make_uint2(l01, l23);
}

// ---- main GEMM: gathered A (LDS-staged, hi/lo split) x pre-split W ----
template <bool PRESPLIT>
__global__ __launch_bounds__(256) void spiral_wmma_kernel(
    const float* __restrict__ x, const float* __restrict__ Wm,
    const unsigned short* __restrict__ whi, const unsigned short* __restrict__ wlo,
    const float* __restrict__ bias, const int* __restrict__ adj,
    const float* __restrict__ zp, float* __restrict__ out)
{
  // double-buffered gathered A tile, hi and lo planes (f16), padded rows
  __shared__ __align__(16) unsigned short lhi[2][16 * kRowStr];
  __shared__ __align__(16) unsigned short llo[2][16 * kRowStr];
  __shared__ float zprow[16];

  const int tid  = threadIdx.x;
  const int lane = tid & 31;
  const int wv   = tid >> 5;          // wave id -> output column tile
  const int hl   = (lane >> 4) & 1;   // upper half-wave
  const int ln16 = lane & 15;
  const int row0 = blockIdx.x << 4;   // 16 rows per workgroup

  // ---- staging assignment: thread -> (row sr, float-octet sf) ----
  const int sr = tid >> 4;            // 0..15
  const int sf = (tid & 15) << 3;     // 0,8,...,120
  const int grow = row0 + sr;
  const int sb = grow / kNpts;        // batch
  const int sn = grow - sb * kNpts;   // point index
  const float* xrowbase = x + (long)sb * kNpts * kF;
  const int* adjrow = adj + sn * kS;
  const int sdst = sr * kRowStr + sf;

  if (tid < 16) {
    const int g2 = row0 + tid;
    zprow[tid] = zp[g2 % kNpts];
  }

  // stage chunk 0
  {
    const int nb = adjrow[0];
    const float4* src = (const float4*)(xrowbase + (long)nb * kF + sf);
    const float4 a0 = src[0], a1 = src[1];
    const float va[8] = {a0.x, a0.y, a0.z, a0.w, a1.x, a1.y, a1.z, a1.w};
    f16 hh[8], ll[8];
    #pragma unroll
    for (int j = 0; j < 8; ++j) {
      hh[j] = (f16)va[j];
      ll[j] = (f16)(va[j] - (float)hh[j]);
    }
    uint4 uh, ul;
    __builtin_memcpy(&uh, hh, 16);
    __builtin_memcpy(&ul, ll, 16);
    *(uint4*)&lhi[0][sdst] = uh;
    *(uint4*)&llo[0][sdst] = ul;
  }
  __syncthreads();

  v8f acc_hh = {}, acc_hl = {}, acc_lh = {};
  const int oc = (wv << 4) + ln16;          // this lane's output channel

  for (int c = 0; c < kNChunk; ++c) {
    const int pb = c & 1;
    // stage next chunk into the other buffer (overlaps with compute below)
    if (c + 1 < kNChunk) {
      const int nb = adjrow[c + 1];
      const float4* src = (const float4*)(xrowbase + (long)nb * kF + sf);
      const float4 a0 = src[0], a1 = src[1];
      const float va[8] = {a0.x, a0.y, a0.z, a0.w, a1.x, a1.y, a1.z, a1.w};
      f16 hh[8], ll[8];
      #pragma unroll
      for (int j = 0; j < 8; ++j) {
        hh[j] = (f16)va[j];
        ll[j] = (f16)(va[j] - (float)hh[j]);
      }
      uint4 uh, ul;
      __builtin_memcpy(&uh, hh, 16);
      __builtin_memcpy(&ul, ll, 16);
      *(uint4*)&lhi[pb ^ 1][sdst] = uh;
      *(uint4*)&llo[pb ^ 1][sdst] = ul;
    }

    // 4 k-iterations of 32 over this 128-K chunk
    #pragma unroll
    for (int it = 0; it < 4; ++it) {
      // --- A operands (contiguous-k 16x32 A, same k order as B):
      // element j, this lane: K = it*32 + 16*hl + j
      const int aoff = ln16 * kRowStr + it * 32 + hl * 16;
      const uint4 ah0 = *(const uint4*)&lhi[pb][aoff];
      const uint4 ah1 = *(const uint4*)&lhi[pb][aoff + 8];
      const uint4 al0 = *(const uint4*)&llo[pb][aoff];
      const uint4 al1 = *(const uint4*)&llo[pb][aoff + 8];
      const v16h ahi = mk16h(ah0, ah1);
      const v16h alo = mk16h(al0, al1);

      // --- B operands (contiguous-k 32x16 B, same k order as A):
      // element j, this lane: K = c*128 + it*32 + 16*hl + j, N = ln16
      v16h bhi, blo;
      if constexpr (PRESPLIT) {
        const long boff = (long)oc * kK + c * kKC + it * 32 + hl * 16;
        const uint4 bh0 = *(const uint4*)(whi + boff);
        const uint4 bh1 = *(const uint4*)(whi + boff + 8);
        const uint4 bl0 = *(const uint4*)(wlo + boff);
        const uint4 bl1 = *(const uint4*)(wlo + boff + 8);
        bhi = mk16h(bh0, bh1);
        blo = mk16h(bl0, bl1);
      } else {
        const float* wr = Wm + (long)oc * kK + c * kKC + it * 32 + hl * 16;
        const float4* bp = (const float4*)wr;
        const float4 b0 = bp[0], b1 = bp[1], b2 = bp[2], b3 = bp[3];
        const float vb[16] = {b0.x,b0.y,b0.z,b0.w, b1.x,b1.y,b1.z,b1.w,
                              b2.x,b2.y,b2.z,b2.w, b3.x,b3.y,b3.z,b3.w};
        #pragma unroll
        for (int j = 0; j < 16; ++j) {
          const f16 h = (f16)vb[j];
          bhi[j] = h;
          blo[j] = (f16)(vb[j] - (float)h);
        }
      }

      // fp32-accurate product via hi/lo split; independent accumulators for ILP
      acc_hh = __builtin_amdgcn_wmma_f32_16x16x32_f16(false, ahi, false, bhi,
                                                      (short)0, acc_hh, false, false);
      acc_hl = __builtin_amdgcn_wmma_f32_16x16x32_f16(false, ahi, false, blo,
                                                      (short)0, acc_hl, false, false);
      acc_lh = __builtin_amdgcn_wmma_f32_16x16x32_f16(false, alo, false, bhi,
                                                      (short)0, acc_lh, false, false);
    }
    __syncthreads();
  }

  // epilogue: combine split products, bias + relu + zero_padding
  // C/D layout: M = e + 8*hl, N = oc
  const float bv = bias[oc];
  #pragma unroll
  for (int e = 0; e < 8; ++e) {
    const int m = e + (hl << 3);
    float v = acc_hh[e] + (acc_hl[e] + acc_lh[e]) + bv;
    v = v > 0.0f ? v : 0.0f;
    v *= zprow[m];
    out[(long)(row0 + m) * kOut + oc] = v;
  }
}

extern "C" void kernel_launch(void* const* d_in, const int* in_sizes, int n_in,
                              void* d_out, int out_size, void* d_ws, size_t ws_size,
                              hipStream_t stream) {
  (void)in_sizes; (void)n_in;
  const float* x   = (const float*)d_in[0];
  const float* Wm  = (const float*)d_in[1];
  const float* b   = (const float*)d_in[2];
  const int*   adj = (const int*)d_in[3];
  const float* zp  = (const float*)d_in[4];
  float* out = (float*)d_out;

  const int rows = out_size / kOut;   // B*N = 96000
  const int grid = rows / 16;         // 6000 workgroups

  const size_t wplane = (size_t)kOut * kK;          // 262144 halves
  if (ws_size >= 2 * wplane * sizeof(unsigned short)) {
    unsigned short* whi = (unsigned short*)d_ws;
    unsigned short* wlo = whi + wplane;
    wsplit_kernel<<<(int)(wplane / 4 / 256), 256, 0, stream>>>(Wm, whi, wlo);
    spiral_wmma_kernel<true><<<grid, 256, 0, stream>>>(x, Wm, whi, wlo, b, adj, zp, out);
  } else {
    spiral_wmma_kernel<false><<<grid, 256, 0, stream>>>(x, Wm, nullptr, nullptr, b, adj, zp, out);
  }
}
